// ChronosVAdvanced_10453950399201
// MI455X (gfx1250) — compile-verified, matrix operand order checked by reading, not run on hardware
//
#include <hip/hip_runtime.h>
#include <hip/hip_bf16.h>

typedef __attribute__((ext_vector_type(16))) __bf16 v16bf;
typedef __attribute__((ext_vector_type(8)))  float  v8f;
typedef __attribute__((ext_vector_type(4)))  unsigned int v4u;

#define D_MODEL 256
#define D_INNER 512
#define NT      16384   // B*T*H*W = 2*8*32*32
#define B_      2
#define T_      8
#define H_      32
#define W_      32
#define EPS     1e-5f

// ---------- helpers ----------
__device__ __forceinline__ unsigned short f2bf(float f) {
    union { float f; unsigned int u; } v; v.f = f;
    unsigned int r = v.u + 0x7FFFu + ((v.u >> 16) & 1u);   // RNE
    return (unsigned short)(r >> 16);
}
__device__ __forceinline__ float silu_f(float x) { return x / (1.f + __expf(-x)); }

// ---------- f32 -> bf16 convert with optional zero column padding ----------
// dst[r, 0:cols_real) = bf16(src[r, 0:cols_real)); dst[r, cols_real:ld_dst) = 0
__global__ void k_cvt_pad(const float* __restrict__ src, int ld_src,
                          unsigned short* __restrict__ dst, int ld_dst,
                          int rows, int cols_real) {
    int i = blockIdx.x * blockDim.x + threadIdx.x;
    if (i >= rows * ld_dst) return;
    int r = i / ld_dst, c = i - r * ld_dst;
    dst[i] = (c < cols_real) ? f2bf(src[(size_t)r * ld_src + c]) : (unsigned short)0;
}

// ---------- WMMA GEMM: C[M,N] (+bias, optional softplus) = A[M,K] @ W[N,K]^T ----------
// A,W bf16 (ushort), C f32. One wave computes a 16 x (16*TN) strip; K%32==0 (pre-padded).
// Fragment halves are K-contiguous per the ISA 7.12.2 layout -> two b128 loads each.
template<int TN>
__global__ void k_gemm_bf16(const unsigned short* __restrict__ A, int lda,
                            const unsigned short* __restrict__ W, int ldw,
                            const float* __restrict__ bias,
                            float* __restrict__ C, int ldc,
                            int M, int N, int K, int act) {
    const int lane = threadIdx.x & 31;
    const int wave = threadIdx.x >> 5;
    const int ntN  = N / (16 * TN);
    const int nTiles = (M >> 4) * ntN;
    const int tile = blockIdx.x * (blockDim.x >> 5) + wave;
    if (tile >= nTiles) return;            // wave-uniform: EXEC stays all-1 for WMMA
    const int tm = tile / ntN, tn = tile % ntN;
    const int row  = (tm << 4) + (lane & 15);
    const int col0 = tn * (16 * TN) + (lane & 15);
    const int hi   = lane >> 4;            // K-half held by lanes 16..31
    const unsigned short* Ap = A + (size_t)row * lda + hi * 8;
    v8f acc[TN];
#pragma unroll
    for (int j = 0; j < TN; ++j) acc[j] = (v8f){};
#pragma unroll 2
    for (int k0 = 0; k0 < K; k0 += 32) {
        union { v4u q[2]; v16bf v; } fa;
        fa.q[0] = *(const v4u*)(Ap + k0);
        fa.q[1] = *(const v4u*)(Ap + k0 + 16);
#pragma unroll
        for (int j = 0; j < TN; ++j) {
            const unsigned short* Wp = W + (size_t)(col0 + j * 16) * ldw + hi * 8 + k0;
            union { v4u q[2]; v16bf v; } fb;
            fb.q[0] = *(const v4u*)(Wp);
            fb.q[1] = *(const v4u*)(Wp + 16);
            acc[j] = __builtin_amdgcn_wmma_f32_16x16x32_bf16(false, fa.v, false, fb.v,
                                                             (short)0, acc[j], false, false);
        }
    }
#pragma unroll
    for (int j = 0; j < TN; ++j) {
        const int col = col0 + j * 16;
        const float bv = bias ? bias[col] : 0.f;
#pragma unroll
        for (int r = 0; r < 8; ++r) {       // D layout: lanes 16..31 hold M = 8..15
            float v = acc[j][r] + bv;
            if (act == 1) v = (v > 20.f) ? v : log1pf(__expf(v));   // softplus
            C[(size_t)((tm << 4) + hi * 8 + r) * ldc + col] = v;
        }
    }
}

// ---------- conditioning: time-embed + text LN/GELU + adaLN -> scale/shift ----------
__global__ void k_cond(const float* __restrict__ t, const float* __restrict__ clip,
                       const float* __restrict__ time_w, const float* __restrict__ time_b,
                       const float* __restrict__ text_w, const float* __restrict__ text_b,
                       const float* __restrict__ ln_g,   const float* __restrict__ ln_b,
                       const float* __restrict__ adaln_w,const float* __restrict__ adaln_b,
                       float* __restrict__ scale_shift) {
    __shared__ float s_te[256], s_sc[256], s_red[256];
    const int b = blockIdx.x, tid = threadIdx.x;
    const float tv = t[b];
    {
        int i = (tid < 128) ? tid : tid - 128;
        float emb = __expf((float)i * (-logf(10000.f) / 127.f));
        float ang = tv * emb;
        s_te[tid] = (tid < 128) ? __sinf(ang) : __cosf(ang);
    }
    __syncthreads();
    float acc = time_b[tid];
    for (int i = 0; i < 256; ++i) acc += s_te[i] * time_w[tid * 256 + i];
    const float t_emb = silu_f(acc);
    acc = text_b[tid];
    for (int i = 0; i < 512; ++i) acc += clip[b * 512 + i] * text_w[tid * 512 + i];
    s_red[tid] = acc; __syncthreads();
    for (int s = 128; s > 0; s >>= 1) { if (tid < s) s_red[tid] += s_red[tid + s]; __syncthreads(); }
    const float mu = s_red[0] / 256.f; __syncthreads();
    const float d = acc - mu;
    s_red[tid] = d * d; __syncthreads();
    for (int s = 128; s > 0; s >>= 1) { if (tid < s) s_red[tid] += s_red[tid + s]; __syncthreads(); }
    const float var = s_red[0] / 256.f;
    const float xn = d * rsqrtf(var + EPS) * ln_g[tid] + ln_b[tid];
    const float g  = 0.5f * xn * (1.f + erff(xn * 0.70710678118f));   // exact GELU
    const float cond = t_emb + g;
    s_sc[tid] = silu_f(cond);
    __syncthreads();
    for (int o = tid; o < 512; o += 256) {
        float m = adaln_b[o];
        for (int i = 0; i < 256; ++i) m += s_sc[i] * adaln_w[o * 256 + i];
        scale_shift[b * 512 + o] = m;
    }
}

// ---------- per-token: VAE proj + adaLN modulation + rope3d + layernorm -> bf16 ----------
__global__ void k_token_pre(const float* __restrict__ vae, const float* __restrict__ vae_w,
                            const float* __restrict__ vae_b, const float* __restrict__ sc_sh,
                            const float* __restrict__ norm_g, const float* __restrict__ norm_b,
                            float* __restrict__ x_mod, unsigned short* __restrict__ xr_bf) {
    __shared__ float sx[256], s_red[256];
    const int tok = blockIdx.x, c = threadIdx.x;
    const int w = tok & 31, h = (tok >> 5) & 31, tt = (tok >> 10) & 7, b = tok >> 13;
    float x = vae_b[c];
#pragma unroll
    for (int j = 0; j < 4; ++j) x += vae[tok * 4 + j] * vae_w[c * 4 + j];
    x = x * (1.f + sc_sh[b * 512 + c]) + sc_sh[b * 512 + 256 + c];
    x_mod[(size_t)tok * 256 + c] = x;
    sx[c] = x; __syncthreads();
    const int chunk = c / 86, local = c - chunk * 86;
    const int pos = (chunk == 0) ? tt : ((chunk == 1) ? h : w);
    const int i = local >> 1;
    const float freq = __powf(10000.f, -(float)(2 * i) / 86.f);
    const float ang = (float)pos * freq;
    const float sn = __sinf(ang), cs = __cosf(ang);
    const int even = !(local & 1);
    const int pidx = even ? (c + 1) : (c - 1);
    const float other = (pidx < 256) ? sx[pidx] : 0.f;
    const float r = even ? (x * cs - other * sn) : (other * sn + x * cs);
    s_red[c] = r; __syncthreads();
    for (int s = 128; s > 0; s >>= 1) { if (c < s) s_red[c] += s_red[c + s]; __syncthreads(); }
    const float mu = s_red[0] / 256.f; __syncthreads();
    s_red[c] = (r - mu) * (r - mu); __syncthreads();
    for (int s = 128; s > 0; s >>= 1) { if (c < s) s_red[c] += s_red[c + s]; __syncthreads(); }
    const float var = s_red[0] / 256.f;
    const float xn = (r - mu) * rsqrtf(var + EPS) * norm_g[c] + norm_b[c];
    xr_bf[(size_t)tok * 256 + c] = f2bf(xn);
}

// ---------- depthwise causal conv (K=4) + silu along one axis ----------
__global__ void k_conv_silu(const float* __restrict__ xz, const float* __restrict__ conv_w,
                            const float* __restrict__ conv_b,
                            float* __restrict__ xs, unsigned short* __restrict__ xs_bf,
                            int L, int stride, int inner, int bigstride) {
    const int s = blockIdx.x;
    const int d = blockIdx.y * 256 + threadIdx.x;
    const size_t base = (size_t)(s / inner) * bigstride + (s % inner);
    const float w0 = conv_w[d * 4 + 0], w1 = conv_w[d * 4 + 1],
                w2 = conv_w[d * 4 + 2], w3 = conv_w[d * 4 + 3];
    const float cb = conv_b[d];
    float x0 = 0.f, x1 = 0.f, x2 = 0.f;
    for (int l = 0; l < L; ++l) {
        const size_t tok = base + (size_t)l * stride;
        const float x3 = xz[tok * 1024 + d];
        const float y = w0 * x0 + w1 * x1 + w2 * x2 + w3 * x3 + cb;
        const float v = silu_f(y);
        xs[tok * 512 + d] = v;
        xs_bf[tok * 512 + d] = f2bf(v);
        x0 = x1; x1 = x2; x2 = x3;
    }
}

// ---------- selective scan + D skip + z-gate -> bf16 ----------
__global__ void k_scan(const float* __restrict__ xs, const float* __restrict__ xz,
                       const float* __restrict__ x_dbl, const float* __restrict__ dt_full,
                       const float* __restrict__ A_log, const float* __restrict__ Dp,
                       unsigned short* __restrict__ yss_bf,
                       int L, int stride, int inner, int bigstride) {
    __shared__ float sB[16], sC[16];
    const int s = blockIdx.x;
    const int d = blockIdx.y * 256 + threadIdx.x;
    const size_t base = (size_t)(s / inner) * bigstride + (s % inner);
    float Anow[16], h[16];
#pragma unroll
    for (int j = 0; j < 16; ++j) { Anow[j] = -__expf(A_log[d * 16 + j]); h[j] = 0.f; }
    const float Dd = Dp[d];
    for (int l = 0; l < L; ++l) {
        const size_t tok = base + (size_t)l * stride;
        if (threadIdx.x < 16)       sB[threadIdx.x]      = x_dbl[tok * 48 + 16 + threadIdx.x];
        else if (threadIdx.x < 32)  sC[threadIdx.x - 16] = x_dbl[tok * 48 + 32 + (threadIdx.x - 16)];
        __syncthreads();
        const float dt = dt_full[tok * 512 + d];
        const float xv = xs[tok * 512 + d];
        float y = 0.f;
#pragma unroll
        for (int j = 0; j < 16; ++j) {
            h[j] = h[j] * __expf(dt * Anow[j]) + dt * sB[j] * xv;
            y += h[j] * sC[j];
        }
        const float yss = y + xv * Dd;
        const float z = xz[tok * 1024 + 512 + d];
        yss_bf[tok * 512 + d] = f2bf(yss * silu_f(z));
        __syncthreads();
    }
}

// ---------- wavelet mask + router + output head (256->4) ----------
__global__ void k_wavelet_out(const float* __restrict__ x_mod, const float* __restrict__ vel,
                              const float* __restrict__ out_w, const float* __restrict__ out_b,
                              const float* __restrict__ gate, float* __restrict__ out) {
    __shared__ float s_red[256];
    __shared__ float routed[4][256];
    const int q = blockIdx.x;
    const int w2 = q & 15, h2 = (q >> 4) & 15, tt = (q >> 8) & 7, b = q >> 11;
    const int c = threadIdx.x;
    const size_t tokA = ((((size_t)b * T_ + tt) * H_ + 2 * h2) * W_ + 2 * w2);
    const size_t toks[4] = { tokA, tokA + 1, tokA + W_, tokA + W_ + 1 };
    const float a  = x_mod[toks[0] * 256 + c];
    const float bb = x_mod[toks[1] * 256 + c];
    const float cc = x_mod[toks[2] * 256 + c];
    const float dd = x_mod[toks[3] * 256 + c];
    const float hl = (-a - bb + cc + dd) * 0.25f;
    const float lh = (-a + bb - cc + dd) * 0.25f;
    const float hh = ( a - bb - cc + dd) * 0.25f;
    s_red[c] = fabsf(hl) + fabsf(lh) + fabsf(hh);
    __syncthreads();
    for (int s = 128; s > 0; s >>= 1) { if (c < s) s_red[c] += s_red[c + s]; __syncthreads(); }
    const float mask = (s_red[0] / (256.f * 3.f) > 0.1f) ? 1.f : 0.f;
    const float g = gate[0];
#pragma unroll
    for (int p = 0; p < 4; ++p) {
        const float v = vel[toks[p] * 256 + c];
        routed[p][c] = v + v * mask * g;
    }
    __syncthreads();
    if (c < 16) {
        const int p = c >> 2, o = c & 3;
        float acc = out_b[o];
        for (int i = 0; i < 256; ++i) acc += routed[p][i] * out_w[o * 256 + i];
        out[toks[p] * 4 + o] = acc;
    }
}

// ---------- host orchestration ----------
extern "C" void kernel_launch(void* const* d_in, const int* in_sizes, int n_in,
                              void* d_out, int out_size, void* d_ws, size_t ws_size,
                              hipStream_t stream) {
    const float* x_vae  = (const float*)d_in[0];
    const float* clip   = (const float*)d_in[1];
    const float* t_in   = (const float*)d_in[2];
    const float* vae_w  = (const float*)d_in[3];
    const float* vae_b  = (const float*)d_in[4];
    const float* time_w = (const float*)d_in[5];
    const float* time_b = (const float*)d_in[6];
    const float* text_w = (const float*)d_in[7];
    const float* text_b = (const float*)d_in[8];
    const float* tln_g  = (const float*)d_in[9];
    const float* tln_b  = (const float*)d_in[10];
    const float* ada_w  = (const float*)d_in[11];
    const float* ada_b  = (const float*)d_in[12];
    const float* norm_g = (const float*)d_in[13];
    const float* norm_b = (const float*)d_in[14];
    const float* fus_w  = (const float*)d_in[42];
    const float* fus_b  = (const float*)d_in[43];
    const float* out_w  = (const float*)d_in[44];
    const float* out_b  = (const float*)d_in[45];
    const float* gate   = (const float*)d_in[46];

    char* wp = (char*)d_ws;
    auto alloc = [&](size_t bytes) { void* r = wp; wp += (bytes + 255) & ~(size_t)255; return r; };

    float*          sc_sh   = (float*)alloc(2 * 512 * 4);
    float*          x_mod   = (float*)alloc((size_t)NT * 256 * 4);
    unsigned short* xr_bf   = (unsigned short*)alloc((size_t)NT * 256 * 2);
    unsigned short* fus_bf  = (unsigned short*)alloc((size_t)256 * 768 * 2);
    unsigned short* inp_bf  = (unsigned short*)alloc((size_t)1024 * 256 * 2);
    unsigned short* xpr_bf  = (unsigned short*)alloc((size_t)48 * 512 * 2);
    unsigned short* dtw_bf  = (unsigned short*)alloc((size_t)512 * 32 * 2);   // K padded 16->32
    unsigned short* opr_bf  = (unsigned short*)alloc((size_t)256 * 512 * 2);
    float*          xz      = (float*)alloc((size_t)NT * 1024 * 4);
    float*          xs      = (float*)alloc((size_t)NT * 512 * 4);
    unsigned short* xs_bf   = (unsigned short*)alloc((size_t)NT * 512 * 2);
    float*          x_dbl   = (float*)alloc((size_t)NT * 48 * 4);
    unsigned short* dtA_bf  = (unsigned short*)alloc((size_t)NT * 32 * 2);    // K padded 16->32
    float*          dt_full = (float*)alloc((size_t)NT * 512 * 4);
    unsigned short* yss_bf  = (unsigned short*)alloc((size_t)NT * 512 * 2);
    float*          cat_f   = (float*)alloc((size_t)NT * 768 * 4);
    unsigned short* cat_bf  = (unsigned short*)alloc((size_t)NT * 768 * 2);
    float*          vel     = (float*)alloc((size_t)NT * 256 * 4);

    auto cvt = [&](const float* src, int ld_src, unsigned short* dst, int ld_dst,
                   int rows, int cols_real) {
        int n = rows * ld_dst;
        k_cvt_pad<<<(n + 255) / 256, 256, 0, stream>>>(src, ld_src, dst, ld_dst, rows, cols_real);
    };
    auto gemm4 = [&](const unsigned short* A, int lda, const unsigned short* W, int ldw,
                     const float* bias, float* C, int ldc, int M, int N, int K, int act) {
        int tiles = (M / 16) * (N / 64);
        k_gemm_bf16<4><<<(tiles + 7) / 8, 256, 0, stream>>>(A, lda, W, ldw, bias, C, ldc, M, N, K, act);
    };
    auto gemm1 = [&](const unsigned short* A, int lda, const unsigned short* W, int ldw,
                     const float* bias, float* C, int ldc, int M, int N, int K, int act) {
        int tiles = (M / 16) * (N / 16);
        k_gemm_bf16<1><<<(tiles + 7) / 8, 256, 0, stream>>>(A, lda, W, ldw, bias, C, ldc, M, N, K, act);
    };

    // conditioning + per-token preprocessing
    k_cond<<<B_, 256, 0, stream>>>(t_in, clip, time_w, time_b, text_w, text_b,
                                   tln_g, tln_b, ada_w, ada_b, sc_sh);
    k_token_pre<<<NT, 256, 0, stream>>>(x_vae, vae_w, vae_b, sc_sh, norm_g, norm_b, x_mod, xr_bf);
    cvt(fus_w, 768, fus_bf, 768, 256, 768);

    // three mamba blocks (T, H, W axes), sequential over shared workspace
    const int Ls[3]      = { T_,           H_,          W_ };
    const int strides[3] = { H_ * W_,      W_,          1  };
    const int inners[3]  = { H_ * W_,      W_,          1  };
    const int bigstr[3]  = { T_ * H_ * W_, H_ * W_,     W_ };
    const int nseqs[3]   = { B_ * H_ * W_, B_ * T_ * W_, B_ * T_ * H_ };

    for (int ax = 0; ax < 3; ++ax) {
        const int pb = 15 + 9 * ax;
        const float* in_proj  = (const float*)d_in[pb + 0];
        const float* conv_w   = (const float*)d_in[pb + 1];
        const float* conv_b   = (const float*)d_in[pb + 2];
        const float* x_proj   = (const float*)d_in[pb + 3];
        const float* dt_w     = (const float*)d_in[pb + 4];
        const float* dt_b     = (const float*)d_in[pb + 5];
        const float* A_log    = (const float*)d_in[pb + 6];
        const float* Dp       = (const float*)d_in[pb + 7];
        const float* out_proj = (const float*)d_in[pb + 8];

        cvt(in_proj, 256,  inp_bf, 256, 1024, 256);
        cvt(x_proj,  512,  xpr_bf, 512, 48,   512);
        cvt(dt_w,    16,   dtw_bf, 32,  512,  16);     // zero-padded K
        cvt(out_proj,512,  opr_bf, 512, 256,  512);

        // xz = xr @ in_proj^T          [NT,1024]
        gemm4(xr_bf, 256, inp_bf, 256, nullptr, xz, 1024, NT, 1024, 256, 0);
        // depthwise causal conv + silu -> xs (f32 + bf16)
        k_conv_silu<<<dim3(nseqs[ax], 2), 256, 0, stream>>>(xz, conv_w, conv_b, xs, xs_bf,
                                                            Ls[ax], strides[ax], inners[ax], bigstr[ax]);
        // x_dbl = xs @ x_proj^T        [NT,48]
        gemm1(xs_bf, 512, xpr_bf, 512, nullptr, x_dbl, 48, NT, 48, 512, 0);
        // dt A-matrix: cols 0..15 of x_dbl, zero-padded to K=32
        cvt(x_dbl, 48, dtA_bf, 32, NT, 16);
        // dt = softplus(dtA @ dt_w^T + dt_b)   [NT,512]
        gemm4(dtA_bf, 32, dtw_bf, 32, dt_b, dt_full, 512, NT, 512, 32, 1);
        // selective scan + gate
        k_scan<<<dim3(nseqs[ax], 2), 256, 0, stream>>>(xs, xz, x_dbl, dt_full, A_log, Dp, yss_bf,
                                                       Ls[ax], strides[ax], inners[ax], bigstr[ax]);
        // out_proj -> its slice of cat  [NT,256] into ld 768
        gemm4(yss_bf, 512, opr_bf, 512, nullptr, cat_f + ax * 256, 768, NT, 256, 512, 0);
    }

    // fusion: velocity = cat @ fusion_w^T + b   [NT,256]
    cvt(cat_f, 768, cat_bf, 768, NT, 768);
    gemm4(cat_bf, 768, fus_bf, 768, fus_b, vel, 256, NT, 256, 768, 0);

    // wavelet mask + router + output head
    k_wavelet_out<<<B_ * T_ * 16 * 16, 256, 0, stream>>>(x_mod, vel, out_w, out_b, gate, (float*)d_out);
}
